// EncoderLayer_49452253447025
// MI455X (gfx1250) — compile-verified, matrix operand order checked by reading, not run on hardware
//
#include <hip/hip_runtime.h>
#include <hip/hip_bf16.h>
#include <math.h>

// Problem constants (from reference)
static constexpr int NB    = 4;
static constexpr int LQ    = 1024;
static constexpr int HD    = 1024;
static constexpr int NHEAD = 16;
static constexpr int HS    = 64;
static constexpr int FFN   = 4096;
static constexpr int MR    = NB * LQ;       // 4096 rows

typedef __attribute__((ext_vector_type(16))) _Float16 v16h;
typedef __attribute__((ext_vector_type(8)))  _Float16 v8h;
typedef __attribute__((ext_vector_type(8)))  float    v8f;
typedef __attribute__((ext_vector_type(4)))  int      v4i;

#define AS1 __attribute__((address_space(1)))
#define AS3 __attribute__((address_space(3)))

// ---- CDNA5 async global->LDS copy (ASYNCcnt path), with safe fallback ------
#if defined(__has_builtin)
#  if __has_builtin(__builtin_amdgcn_global_load_async_to_lds_b128)
#    define HAVE_ASYNC_LDS 1
#  endif
#endif
#ifndef HAVE_ASYNC_LDS
#  define HAVE_ASYNC_LDS 0
#endif

// Copy 16 bytes (8 halfs) from global to LDS, per lane.
__device__ __forceinline__ void cp16(const _Float16* gp, _Float16* lp) {
#if HAVE_ASYNC_LDS
    __builtin_amdgcn_global_load_async_to_lds_b128(
        (AS1 v4i*)(AS1 void*)gp, (AS3 v4i*)(AS3 void*)lp, 0, 0);
#else
    *(v8h*)lp = *(const v8h*)gp;
#endif
}

__device__ __forceinline__ void cp16_wait() {
#if HAVE_ASYNC_LDS
#  if defined(__has_builtin) && __has_builtin(__builtin_amdgcn_s_wait_asynccnt)
    __builtin_amdgcn_s_wait_asynccnt(0);
#  else
    asm volatile("s_wait_asynccnt 0x0" ::: "memory");
#  endif
#endif
}

// Load a 16x32 f16 fragment (A or B operand) from LDS laid out [row][k],
// k contiguous with `stride` halfs (multiple of 8 -> rows 16B aligned).
// CDNA5 16-bit operand layout:
//   lanes 0-15 : row = lane,    regs hold K = 0..7 and 16..23
//   lanes 16-31: row = lane-16, regs hold K = 8..15 and 24..31
__device__ __forceinline__ v16h frag_ld(const _Float16* tile, int stride, int lane) {
    const int r  = lane & 15;
    const int hi = (lane >> 4) & 1;
    const _Float16* p = tile + r * stride + hi * 8;
    v8h lo = *(const v8h*)(p);
    v8h hh = *(const v8h*)(p + 16);
    v16h out;
#pragma unroll
    for (int i = 0; i < 8; ++i) { out[i] = lo[i]; out[8 + i] = hh[i]; }
    return out;
}

// ---------------------------------------------------------------------------
// f32 -> f16 elementwise convert (x), 4 elems/thread
// ---------------------------------------------------------------------------
__global__ __launch_bounds__(256)
void conv_h(const float* __restrict__ in, _Float16* __restrict__ out) {
    size_t i0 = ((size_t)blockIdx.x * 256 + threadIdx.x) * 4;
#pragma unroll
    for (int i = 0; i < 4; ++i) out[i0 + i] = (_Float16)in[i0 + i];
}

// ---------------------------------------------------------------------------
// Weight convert + transpose: W f32 [K][Nd] -> Wt f16 [Nd][K]  (32x32 tiles)
// ---------------------------------------------------------------------------
__global__ __launch_bounds__(256)
void conv_t(const float* __restrict__ W, _Float16* __restrict__ Wt, int K, int Nd) {
    __shared__ float T[32][33];
    const int tid = threadIdx.x;
    const int bk = blockIdx.y * 32, bn = blockIdx.x * 32;
    const int c = tid & 31, r8 = tid >> 5;
#pragma unroll
    for (int p = 0; p < 4; ++p)
        T[r8 + p * 8][c] = W[(size_t)(bk + r8 + p * 8) * Nd + bn + c];
    __syncthreads();
#pragma unroll
    for (int p = 0; p < 4; ++p)
        Wt[(size_t)(bn + r8 + p * 8) * K + bk + c] = (_Float16)T[c][r8 + p * 8];
}

// ---------------------------------------------------------------------------
// Tiled GEMM: C[M,Nd](f16) = act(A[M,K](f16) @ Bt[Nd,K](f16)^T + bias[Nd](f32))
// Block: 256 threads (8 waves, 4x2), tile 128x128, wave tile 32x64, K-step 32.
// Double-buffered LDS: async copies for tile k+1 are issued before computing
// tile k, so the ASYNCcnt wait at the top of the next iteration is hidden by
// a full WMMA phase. Assumes M%128==0, Nd%128==0, K%32==0.
// ---------------------------------------------------------------------------
template <bool RELU>
__global__ __launch_bounds__(256)
void gemm_h(const _Float16* __restrict__ A, const _Float16* __restrict__ Bt,
            const float* __restrict__ bias, _Float16* __restrict__ C,
            int M, int K, int Nd) {
    __shared__ _Float16 As[2][128 * 40];   // [m][k], stride 40 halfs (80B)
    __shared__ _Float16 Bs[2][128 * 40];   // [n][k]

    const int tid  = threadIdx.x;
    const int lane = tid & 31;
    const int wave = tid >> 5;
    const int wm   = wave >> 1;            // 0..3 (32 rows each)
    const int wn   = wave & 1;             // 0..1 (64 cols each)
    const int bm   = blockIdx.y * 128;
    const int bn   = blockIdx.x * 128;

    // Issue one 128x32 A tile + 128x32 B tile into buffer p (16B async chunks)
    auto issue_tile = [&](int k0, int p) {
#pragma unroll
        for (int i = 0; i < 2; ++i) {
            int cI = tid + i * 256;              // 0..511
            int ck = cI & 3, mm = cI >> 2;
            cp16(&A[(size_t)(bm + mm) * K + k0 + ck * 8], &As[p][mm * 40 + ck * 8]);
        }
#pragma unroll
        for (int i = 0; i < 2; ++i) {
            int cI = tid + i * 256;              // 0..511
            int ck = cI & 3, nn = cI >> 2;
            cp16(&Bt[(size_t)(bn + nn) * K + k0 + ck * 8], &Bs[p][nn * 40 + ck * 8]);
        }
    };

    v8f acc[2][4] = {};

    issue_tile(0, 0);
    for (int k0 = 0; k0 < K; k0 += 32) {
        const int p = (k0 >> 5) & 1;
        cp16_wait();            // my copies into buffer p have landed
        __syncthreads();        // everyone's copies landed; buffer p^1 free
        if (k0 + 32 < K) issue_tile(k0 + 32, p ^ 1);

        v16h af[2], bf[4];
#pragma unroll
        for (int mi = 0; mi < 2; ++mi)
            af[mi] = frag_ld(&As[p][(wm * 32 + mi * 16) * 40], 40, lane);
#pragma unroll
        for (int ni = 0; ni < 4; ++ni)
            bf[ni] = frag_ld(&Bs[p][(wn * 64 + ni * 16) * 40], 40, lane);
#pragma unroll
        for (int mi = 0; mi < 2; ++mi)
#pragma unroll
            for (int ni = 0; ni < 4; ++ni)
                acc[mi][ni] = __builtin_amdgcn_wmma_f32_16x16x32_f16(
                    false, af[mi], false, bf[ni], (short)0, acc[mi][ni], false, false);
    }

    // C layout: row m = r + 8*(lane>=16), col n = lane&15
    const int l15 = lane & 15, hi = lane >> 4;
#pragma unroll
    for (int mi = 0; mi < 2; ++mi)
#pragma unroll
        for (int ni = 0; ni < 4; ++ni) {
            const int gcol = bn + wn * 64 + ni * 16 + l15;
            const float bv = bias[gcol];
#pragma unroll
            for (int r = 0; r < 8; ++r) {
                int grow = bm + wm * 32 + mi * 16 + r + hi * 8;
                float v = acc[mi][ni][r] + bv;
                if (RELU) v = fmaxf(v, 0.0f);
                C[(size_t)grow * Nd + gcol] = (_Float16)v;
            }
        }
}

// ---------------------------------------------------------------------------
// Flash attention: one block = (batch n, head h, 128 query rows), f16 in/out.
// 8 waves; each wave owns 16 query rows; key blocks of 64; online softmax.
// ---------------------------------------------------------------------------
__global__ __launch_bounds__(256)
void flash_attn(const _Float16* __restrict__ Q, const _Float16* __restrict__ Km,
                const _Float16* __restrict__ V, const long long* __restrict__ mask,
                _Float16* __restrict__ Ctx) {
    __shared__ _Float16 Qs[128 * 72];        // [qrow][s]
    __shared__ _Float16 Ks[64 * 72];         // [key][s]   (B of S-GEMM)
    __shared__ _Float16 Vs[64 * 72];         // [s][key]   (B of O-GEMM)
    __shared__ _Float16 Ps[8 * 16 * 72];     // wave-private P staging [row][key]

    const int tid  = threadIdx.x;
    const int lane = tid & 31;
    const int wave = tid >> 5;
    const int n    = blockIdx.z;
    const int h    = blockIdx.y;
    const int q0   = blockIdx.x * 128;
    const size_t hoff = (size_t)h * HS;

    // Q block: 128 rows x 8 chunks = 1024 chunks, 4/thread (async)
#pragma unroll
    for (int i = 0; i < 4; ++i) {
        int cI = tid + i * 256;              // 0..1023
        int ck = cI & 7, rr = cI >> 3;
        cp16(&Q[((size_t)(n * LQ + q0 + rr)) * HD + hoff + ck * 8],
             &Qs[rr * 72 + ck * 8]);
    }

    const int l15 = lane & 15, hi = lane >> 4;
    float rm[8], rl[8];
#pragma unroll
    for (int r = 0; r < 8; ++r) { rm[r] = -INFINITY; rl[r] = 0.0f; }
    v8f oacc[4] = {};

    _Float16* Pw = &Ps[wave * 16 * 72];

    for (int kb = 0; kb < LQ; kb += 64) {
        __syncthreads();
        // K tile: 64 rows x 8 chunks = 512 chunks, 2/thread (async)
#pragma unroll
        for (int i = 0; i < 2; ++i) {
            int cI = tid + i * 256;
            int ck = cI & 7, kr = cI >> 3;
            cp16(&Km[((size_t)(n * LQ + kb + kr)) * HD + hoff + ck * 8],
                 &Ks[kr * 72 + ck * 8]);
        }
        // V tile transposed into [s][key]: VGPR path (genuine transpose)
#pragma unroll
        for (int i = 0; i < 2; ++i) {
            int cI = tid + i * 256;
            int ck = cI & 7, kr = cI >> 3;
            v8h vv = *(const v8h*)&V[((size_t)(n * LQ + kb + kr)) * HD + hoff + ck * 8];
#pragma unroll
            for (int j = 0; j < 8; ++j) Vs[(ck * 8 + j) * 72 + kr] = vv[j];
        }
        cp16_wait();           // covers Q (first iter) + K asyncs of this wave
        __syncthreads();

        // S = Q_wave(16x64) @ K_blk^T : 4 key tiles, 2 K-steps
        v8f sc[4] = {};
#pragma unroll
        for (int ks = 0; ks < 64; ks += 32) {
            v16h aq = frag_ld(&Qs[(wave * 16) * 72 + ks], 72, lane);
#pragma unroll
            for (int t = 0; t < 4; ++t) {
                v16h bk = frag_ld(&Ks[(t * 16) * 72 + ks], 72, lane);
                sc[t] = __builtin_amdgcn_wmma_f32_16x16x32_f16(
                    false, aq, false, bk, (short)0, sc[t], false, false);
            }
        }

        // Scale + key padding mask (this lane's column = kb + t*16 + l15)
#pragma unroll
        for (int t = 0; t < 4; ++t) {
            long long mv = mask[(size_t)n * LQ + kb + t * 16 + l15];
            bool dead = (mv == 0);
#pragma unroll
            for (int r = 0; r < 8; ++r) {
                float v = sc[t][r] * 0.125f;          // 1/sqrt(64)
                sc[t][r] = dead ? -INFINITY : v;
            }
        }

        // Tile row max (reduce across the 16 lanes of each half)
        float tm[8];
#pragma unroll
        for (int r = 0; r < 8; ++r) {
            float v = sc[0][r];
#pragma unroll
            for (int t = 1; t < 4; ++t) v = fmaxf(v, sc[t][r]);
#pragma unroll
            for (int m = 1; m < 16; m <<= 1) v = fmaxf(v, __shfl_xor(v, m, 32));
            tm[r] = v;
        }

        // Online rescale
        float pscale[8];
#pragma unroll
        for (int r = 0; r < 8; ++r) {
            float nm = fmaxf(rm[r], tm[r]);
            pscale[r] = (nm == -INFINITY) ? 1.0f : __expf(rm[r] - nm);
            rm[r] = nm;
        }

        // P = exp(S - rowmax) -> LDS (A layout [row][key]); accumulate row sums
        float tsum[8];
#pragma unroll
        for (int r = 0; r < 8; ++r) tsum[r] = 0.0f;
#pragma unroll
        for (int t = 0; t < 4; ++t)
#pragma unroll
            for (int r = 0; r < 8; ++r) {
                float e = (rm[r] == -INFINITY) ? 0.0f : __expf(sc[t][r] - rm[r]);
                Pw[(r + 8 * hi) * 72 + t * 16 + l15] = (_Float16)e;
                tsum[r] += e;
            }
#pragma unroll
        for (int r = 0; r < 8; ++r) {
            float s = tsum[r];
#pragma unroll
            for (int m = 1; m < 16; m <<= 1) s += __shfl_xor(s, m, 32);
            rl[r] = rl[r] * pscale[r] + s;
        }

        // Rescale O accumulators (same row<->reg mapping as S)
#pragma unroll
        for (int t = 0; t < 4; ++t)
#pragma unroll
            for (int r = 0; r < 8; ++r) oacc[t][r] *= pscale[r];

        // O += P(16x64) @ V_blk(64x64). Pw is wave-private; per-wave LDS ops
        // are in-order, so no barrier between store and load.
#pragma unroll
        for (int ks = 0; ks < 64; ks += 32) {
            v16h ap = frag_ld(Pw + ks, 72, lane);
#pragma unroll
            for (int t = 0; t < 4; ++t) {
                v16h bv = frag_ld(&Vs[(t * 16) * 72 + ks], 72, lane);
                oacc[t] = __builtin_amdgcn_wmma_f32_16x16x32_f16(
                    false, ap, false, bv, (short)0, oacc[t], false, false);
            }
        }
    }

    // Normalize and write ctx (n, q, h*HS + s) as f16
#pragma unroll
    for (int t = 0; t < 4; ++t)
#pragma unroll
        for (int r = 0; r < 8; ++r) {
            int qrow = q0 + wave * 16 + r + 8 * hi;
            int s    = t * 16 + l15;
            float d  = rl[r];
            float v  = (d > 0.0f) ? oacc[t][r] / d : 0.0f;
            Ctx[((size_t)(n * LQ + qrow)) * HD + hoff + s] = (_Float16)v;
        }
}

// ---------------------------------------------------------------------------
// Row LayerNorm + post-LN residual (cols = 1024):
//   OutF = LN(X)*g + b + Res (f32);  OutH = f16 copy of OutF (optional)
// ---------------------------------------------------------------------------
__global__ __launch_bounds__(256)
void ln_residual(const _Float16* __restrict__ X, const float* __restrict__ g,
                 const float* __restrict__ b, const float* __restrict__ Res,
                 float* __restrict__ OutF, _Float16* __restrict__ OutH, int cols) {
    __shared__ float red[256];
    const int tid = threadIdx.x;
    const int row = blockIdx.x;
    const _Float16* xr = X + (size_t)row * cols;

    float s = 0.0f;
    for (int c = tid; c < cols; c += 256) s += (float)xr[c];
    red[tid] = s; __syncthreads();
    for (int off = 128; off > 0; off >>= 1) {
        if (tid < off) red[tid] += red[tid + off];
        __syncthreads();
    }
    float mean = red[0] / cols;
    __syncthreads();

    float vs = 0.0f;
    for (int c = tid; c < cols; c += 256) { float d = (float)xr[c] - mean; vs += d * d; }
    red[tid] = vs; __syncthreads();
    for (int off = 128; off > 0; off >>= 1) {
        if (tid < off) red[tid] += red[tid + off];
        __syncthreads();
    }
    float inv = rsqrtf(red[0] / cols + 1e-5f);

    for (int c = tid; c < cols; c += 256) {
        size_t i = (size_t)row * cols + c;
        float v = ((float)xr[c] - mean) * inv * g[c] + b[c] + Res[i];
        OutF[i] = v;
        if (OutH) OutH[i] = (_Float16)v;
    }
}

// ---------------------------------------------------------------------------
extern "C" void kernel_launch(void* const* d_in, const int* in_sizes, int n_in,
                              void* d_out, int out_size, void* d_ws, size_t ws_size,
                              hipStream_t stream) {
    const float* x    = (const float*)d_in[0];
    const long long* mask = (const long long*)d_in[1];
    const float* Wq = (const float*)d_in[2];  const float* bq = (const float*)d_in[3];
    const float* Wk = (const float*)d_in[4];  const float* bk = (const float*)d_in[5];
    const float* Wv = (const float*)d_in[6];  const float* bv = (const float*)d_in[7];
    const float* Wo = (const float*)d_in[8];  const float* bo = (const float*)d_in[9];
    const float* ln1g = (const float*)d_in[10]; const float* ln1b = (const float*)d_in[11];
    const float* W1 = (const float*)d_in[12]; const float* b1 = (const float*)d_in[13];
    const float* W2 = (const float*)d_in[14]; const float* b2 = (const float*)d_in[15];
    const float* ln2g = (const float*)d_in[16]; const float* ln2b = (const float*)d_in[17];
    float* out = (float*)d_out;

    // Workspace carve-up (all f16 except x1f). Dead buffers are reused.
    char* p = (char*)d_ws;
    auto allocH = [&](size_t n) { _Float16* r = (_Float16*)p; p += n * sizeof(_Float16); return r; };
    const size_t EH = (size_t)MR * HD;                  // 4M elements

    _Float16* xh   = allocH(EH);
    _Float16* Wqt  = allocH((size_t)HD * HD);
    _Float16* Wkt  = allocH((size_t)HD * HD);
    _Float16* Wvt  = allocH((size_t)HD * HD);
    _Float16* Wot  = allocH((size_t)HD * HD);
    _Float16* W1t  = allocH((size_t)HD * FFN);
    _Float16* W2t  = allocH((size_t)FFN * HD);
    _Float16* qh   = allocH(EH);
    _Float16* kh   = allocH(EH);
    _Float16* vh   = allocH(EH);
    _Float16* ctxh = allocH(EH);
    _Float16* ffn1 = allocH((size_t)MR * FFN);
    float*    x1f  = (float*)p; p += EH * sizeof(float);
    _Float16* attno = qh;   // q dead after attention
    _Float16* x1h   = kh;   // k dead after attention
    _Float16* h2h   = vh;   // v dead after attention

    dim3 blk(256);

    // 0) One-time precision conversion: x -> f16; weights -> f16, [Nd][K]
    conv_h<<<dim3(EH / 1024), blk, 0, stream>>>(x, xh);
    dim3 gT (HD / 32, HD / 32);
    conv_t<<<gT, blk, 0, stream>>>(Wq, Wqt, HD, HD);
    conv_t<<<gT, blk, 0, stream>>>(Wk, Wkt, HD, HD);
    conv_t<<<gT, blk, 0, stream>>>(Wv, Wvt, HD, HD);
    conv_t<<<gT, blk, 0, stream>>>(Wo, Wot, HD, HD);
    conv_t<<<dim3(FFN / 32, HD / 32), blk, 0, stream>>>(W1, W1t, HD, FFN);
    conv_t<<<dim3(HD / 32, FFN / 32), blk, 0, stream>>>(W2, W2t, FFN, HD);

    dim3 gH (HD  / 128, MR / 128);
    dim3 gF (FFN / 128, MR / 128);

    // 1) QKV projections (f16 WMMA, double-buffered async LDS staging)
    gemm_h<false><<<gH, blk, 0, stream>>>(xh, Wqt, bq, qh, MR, HD, HD);
    gemm_h<false><<<gH, blk, 0, stream>>>(xh, Wkt, bk, kh, MR, HD, HD);
    gemm_h<false><<<gH, blk, 0, stream>>>(xh, Wvt, bv, vh, MR, HD, HD);

    // 2) Flash attention
    dim3 gA(LQ / 128, NHEAD, NB);
    flash_attn<<<gA, blk, 0, stream>>>(qh, kh, vh, mask, ctxh);

    // 3) Output projection
    gemm_h<false><<<gH, blk, 0, stream>>>(ctxh, Wot, bo, attno, MR, HD, HD);

    // 4) x1 = LN1(attn_o) + x   (f32 chain + f16 operand copy)
    ln_residual<<<dim3(MR), blk, 0, stream>>>(attno, ln1g, ln1b, x, x1f, x1h, HD);

    // 5) FFN
    gemm_h<true ><<<gF, blk, 0, stream>>>(x1h, W1t, b1, ffn1, MR, HD, FFN);
    gemm_h<false><<<gH, blk, 0, stream>>>(ffn1, W2t, b2, h2h, MR, FFN, HD);

    // 6) out = LN2(h2) + x1
    ln_residual<<<dim3(MR), blk, 0, stream>>>(h2h, ln2g, ln2b, x1f, out, nullptr, HD);
}
